// PointNetSetAbstractionMsg_49237505082102
// MI455X (gfx1250) — compile-verified, matrix operand order checked by reading, not run on hardware
//
#include <hip/hip_runtime.h>
#include <hip/hip_bf16.h>
#include <hip/hip_fp16.h>
#include <stdint.h>

typedef _Float16 half_t;
typedef _Float16 v16h __attribute__((ext_vector_type(16)));
typedef float    v8f  __attribute__((ext_vector_type(8)));

#define NPTS   8192
#define BATCH  8
#define SOUT   512
#define OUTCH  320
#define BN_EPS 1e-5f

// K index inside a 32-wide K-slab for the 16-bit A/B WMMA lane layout:
// vector element i (=vgpr*2+half), hi = lane>>4.
__device__ __forceinline__ int kmap(int i, int hi) {
    return ((i >> 3) << 4) + (i & 7) + (hi << 3);
}

// ---------------------------------------------------------------------------
// Fold BN (eval) + bias into f16 weights (cin zero-padded to cinp) + f32 bias.
// ---------------------------------------------------------------------------
__global__ void fold_layer_kernel(const float* __restrict__ W, const float* __restrict__ b,
                                  const float* __restrict__ gamma, const float* __restrict__ beta,
                                  const float* __restrict__ mu, const float* __restrict__ var,
                                  int cout, int cin, int cinp,
                                  half_t* __restrict__ Wp, float* __restrict__ bp) {
    int idx = blockIdx.x * blockDim.x + threadIdx.x;
    int total = cout * cinp;
    if (idx < total) {
        int co = idx / cinp, ci = idx - co * cinp;
        float s = gamma[co] * rsqrtf(var[co] + BN_EPS);
        float w = (ci < cin) ? W[co * cin + ci] * s : 0.0f;
        Wp[idx] = (half_t)w;
    }
    if (idx < cout) {
        float s = gamma[idx] * rsqrtf(var[idx] + BN_EPS);
        bp[idx] = (b[idx] - mu[idx]) * s + beta[idx];
    }
}

// ---------------------------------------------------------------------------
// Farthest point sampling: one block per batch. dist[] lives in LDS.
// Matches jnp.argmax first-occurrence tie-break (smaller index wins on ties).
// ---------------------------------------------------------------------------
__global__ void __launch_bounds__(256)
fps_kernel(const float* __restrict__ xyz, int* __restrict__ fps_idx,
           float* __restrict__ new_xyz, float* __restrict__ out_xyz_t) {
    const int b = blockIdx.x;
    const int tid = threadIdx.x;
    constexpr int T = 256;
    __shared__ float dist[NPTS];
    __shared__ float redV[T];
    __shared__ int   redI[T];
    __shared__ int   curS;

    const float* X = xyz + (size_t)b * 3 * NPTS;
    for (int j = tid; j < NPTS; j += T) dist[j] = 3.4e38f;
    if (tid == 0) curS = 0;
    __syncthreads();

    for (int s = 0; s < SOUT; ++s) {
        const int cur = curS;
        const float px = X[cur];
        const float py = X[NPTS + cur];
        const float pz = X[2 * NPTS + cur];
        if (tid == 0) {
            fps_idx[b * SOUT + s] = cur;
            float* nx = new_xyz + (((size_t)b * SOUT) + s) * 3;
            nx[0] = px; nx[1] = py; nx[2] = pz;
            out_xyz_t[((size_t)b * 3 + 0) * SOUT + s] = px;
            out_xyz_t[((size_t)b * 3 + 1) * SOUT + s] = py;
            out_xyz_t[((size_t)b * 3 + 2) * SOUT + s] = pz;
        }
        float bv = -1.0f; int bi = NPTS;
        for (int j = tid; j < NPTS; j += T) {
            float dx = X[j] - px, dy = X[NPTS + j] - py, dz = X[2 * NPTS + j] - pz;
            float d = dx * dx + dy * dy + dz * dz;
            float nd = fminf(dist[j], d);
            dist[j] = nd;
            if (nd > bv) { bv = nd; bi = j; }   // strict > keeps first occurrence
        }
        redV[tid] = bv; redI[tid] = bi;
        __syncthreads();
        for (int off = T / 2; off > 0; off >>= 1) {
            if (tid < off) {
                float v2 = redV[tid + off]; int i2 = redI[tid + off];
                if (v2 > redV[tid] || (v2 == redV[tid] && i2 < redI[tid])) {
                    redV[tid] = v2; redI[tid] = i2;
                }
            }
            __syncthreads();
        }
        if (tid == 0) curS = redI[0];
        __syncthreads();
    }
}

// ---------------------------------------------------------------------------
// One 16x16 WMMA-tiled MLP layer: out = relu(in * W^T + bias), f16 in/out LDS,
// f32 accumulate via v_wmma_f32_16x16x32_f16. Tiles split across 4 waves.
// Both A and B fragments are two 16B runs per lane (k 8h..8h+7, 16+8h..16+8h+7) -> 2x ds_load_b128.
// ---------------------------------------------------------------------------
template<int KS>
__device__ __forceinline__ void mlp_layer(const half_t* __restrict__ inF, int ldIn,
                                          half_t* __restrict__ outF, int ldOut,
                                          const half_t* __restrict__ W,
                                          const float* __restrict__ Bias,
                                          int cinp, int cout, int wave, int lane) {
    const int MT  = KS / 16;
    const int NT  = cout / 16;
    const int col = lane & 15;   // A: M row / B: N col / D: N col
    const int hi  = lane >> 4;
    for (int t = wave; t < MT * NT; t += 4) {
        const int m0 = (t / NT) * 16;
        const int n0 = (t % NT) * 16;
        v8f acc = {0.f, 0.f, 0.f, 0.f, 0.f, 0.f, 0.f, 0.f};
        for (int k0 = 0; k0 < cinp; k0 += 32) {
            v16h a, bm;
#pragma unroll
            for (int i = 0; i < 16; ++i) {
                const int kk = kmap(i, hi);
                a[i]  = inF[(m0 + col) * ldIn + k0 + kk];
                bm[i] = W[(n0 + col) * cinp + k0 + kk];
            }
            acc = __builtin_amdgcn_wmma_f32_16x16x32_f16(
                /*neg_a=*/false, a, /*neg_b=*/false, bm,
                /*c_mod=*/(short)0, acc, /*reuse_a=*/false, /*reuse_b=*/false);
        }
        const float bv = Bias[n0 + col];
#pragma unroll
        for (int v = 0; v < 8; ++v) {
            float x = acc[v] + bv;
            x = x > 0.f ? x : 0.f;
            outF[(m0 + v + hi * 8) * ldOut + (n0 + col)] = (half_t)x;
        }
    }
}

// ---------------------------------------------------------------------------
// Per-scale fused kernel: async-stage folded weights into LDS, ordered first-K
// ball query -> gather into LDS f16 tile -> 3-layer WMMA MLP -> max over K.
// One 128-thread block (4 wave32) per (b, s) group.
// ---------------------------------------------------------------------------
template<int KS, int CO1, int CO2, int CO3, int ACOLS, int BCOLS>
__global__ void __launch_bounds__(128)
msg_group_kernel(const float* __restrict__ xyz, const float* __restrict__ pts,
                 const float* __restrict__ new_xyz,
                 const half_t* __restrict__ Wall,   // W1|W2|W3 contiguous, f16
                 const float* __restrict__ B1, const float* __restrict__ B2,
                 const float* __restrict__ B3,
                 float radius, float* __restrict__ out_pts, int c_off) {
    constexpr int TPB = 128;
    constexpr int WL1 = 32 * CO1;
    constexpr int WL2 = CO1 * CO2;
    constexpr int WL3 = CO2 * CO3;
    constexpr int WHALVES = WL1 + WL2 + WL3;   // bytes = 2*WHALVES, multiple of 2048
    const int g    = blockIdx.x;
    const int b    = g / SOUT;
    const int s    = g - b * SOUT;
    const int tid  = threadIdx.x;
    const int lane = tid & 31;
    const int wave = tid >> 5;

    __shared__ half_t featA[KS * ACOLS];
    __shared__ half_t featB[KS * BCOLS];
    __shared__ half_t wlds[WHALVES];
    __shared__ int    cntS;

    const float* X = xyz + (size_t)b * 3 * NPTS;
    const float* P = pts + (size_t)b * 3 * NPTS;
    const float* C = new_xyz + (((size_t)b * SOUT) + s) * 3;
    const float cx = C[0], cy = C[1], cz = C[2];

    if (tid == 0) {   // warm input lines (global_prefetch_b8)
        __builtin_prefetch(X, 0, 0);
        __builtin_prefetch(P, 0, 0);
    }

    // ---- Async-copy all folded weights for this scale into LDS (ASYNCcnt) ----
    {
        const uint32_t wbase = (uint32_t)(uintptr_t)(void*)wlds;   // LDS byte offset
        const uint64_t wsrc  = (uint64_t)(uintptr_t)(const void*)Wall;
        for (int off = tid * 16; off < WHALVES * 2; off += TPB * 16) {
            uint32_t dst = wbase + (uint32_t)off;
            uint64_t src = wsrc + (uint64_t)off;
            asm volatile("global_load_async_to_lds_b128 %0, %1, off"
                         :: "v"(dst), "v"(src) : "memory");
        }
        asm volatile("s_wait_asynccnt 0x0" ::: "memory");
    }

    // Zero cols [0,32) of the A tile (covers cin=6 padded to 32).
    for (int i = tid; i < KS * 32; i += TPB)
        featA[(i >> 5) * ACOLS + (i & 31)] = (half_t)0.0f;
    __syncthreads();   // weights staged + A tile zeroed, visible to all waves

    // Ordered first-K ball query (pytorch3d semantics), wave 0 only.
    if (wave == 0) {
        int cnt = 0;
        const float r2 = radius * radius;
        for (int base = 0; base < NPTS && cnt < KS; base += 32) {
            const int j = base + lane;
            const float xj = X[j], yj = X[NPTS + j], zj = X[2 * NPTS + j];
            const float dx = xj - cx, dy = yj - cy, dz = zj - cz;
            const bool ok = (dx * dx + dy * dy + dz * dz) < r2;
            const unsigned m = __builtin_amdgcn_ballot_w32(ok);
            const int rank = __popc(m & ((1u << lane) - 1u));
            const int pos = cnt + rank;
            if (ok && pos < KS) {
                half_t* row = &featA[pos * ACOLS];
                row[0] = (half_t)P[j];
                row[1] = (half_t)P[NPTS + j];
                row[2] = (half_t)P[2 * NPTS + j];
                row[3] = (half_t)dx; row[4] = (half_t)dy; row[5] = (half_t)dz;
            }
            cnt += __popc(m);
        }
        if (lane == 0) cntS = (cnt > KS) ? KS : cnt;
    }
    __syncthreads();

    // Pad rows: idx=-1 wraps to N-1 for point feats; grouped xyz = 0 - center.
    for (int r = cntS + tid; r < KS; r += TPB) {
        half_t* row = &featA[r * ACOLS];
        row[0] = (half_t)P[NPTS - 1];
        row[1] = (half_t)P[2 * NPTS - 1];
        row[2] = (half_t)P[3 * NPTS - 1];
        row[3] = (half_t)(-cx); row[4] = (half_t)(-cy); row[5] = (half_t)(-cz);
    }
    __syncthreads();

    mlp_layer<KS>(featA, ACOLS, featB, BCOLS, wlds,             B1, 32,  CO1, wave, lane);
    __syncthreads();
    mlp_layer<KS>(featB, BCOLS, featA, ACOLS, wlds + WL1,       B2, CO1, CO2, wave, lane);
    __syncthreads();
    mlp_layer<KS>(featA, ACOLS, featB, BCOLS, wlds + WL1 + WL2, B3, CO2, CO3, wave, lane);
    __syncthreads();

    // Max-pool over the K samples, write [B, OUTCH, S] slice.
    for (int c = tid; c < CO3; c += TPB) {
        float mv = -3.4e38f;
        for (int r = 0; r < KS; ++r)
            mv = fmaxf(mv, (float)featB[r * BCOLS + c]);
        out_pts[(((size_t)b) * OUTCH + (c_off + c)) * SOUT + s] = mv;
    }
}

// ---------------------------------------------------------------------------
extern "C" void kernel_launch(void* const* d_in, const int* in_sizes, int n_in,
                              void* d_out, int out_size, void* d_ws, size_t ws_size,
                              hipStream_t stream) {
    (void)in_sizes; (void)n_in; (void)out_size; (void)ws_size;
    const float* xyz = (const float*)d_in[0];
    const float* pts = (const float*)d_in[1];
    auto PARAM = [&](int scale, int layer, int t) -> const float* {
        return (const float*)d_in[2 + (scale * 3 + layer) * 6 + t];
    };

    // Workspace layout
    char* ws = (char*)d_ws;
    int*    fps_idx = (int*)ws;                              // 8*512*4      = 16384 B
    float*  nxyz    = (float*)(ws + 16384);                  // 8*512*3*4    = 49152 B
    float*  biasWs  = (float*)(ws + 16384 + 49152);          // 672*4        =  2688 B
    half_t* wWs     = (half_t*)(ws + 16384 + 49152 + 4096);  // 38912 halves = 77824 B

    const int cins [3][3] = {{6, 32, 32}, {6, 64, 64}, {6, 64, 96}};
    const int cinps[3][3] = {{32, 32, 32}, {32, 64, 64}, {32, 64, 96}};
    const int couts[3][3] = {{32, 32, 64}, {64, 64, 128}, {64, 96, 128}};

    half_t* Wp[3][3];
    float*  Bp[3][3];
    size_t woff = 0, boff = 0;
    for (int sc = 0; sc < 3; ++sc)
        for (int l = 0; l < 3; ++l) {
            Wp[sc][l] = wWs + woff;   woff += (size_t)couts[sc][l] * cinps[sc][l];
            Bp[sc][l] = biasWs + boff; boff += couts[sc][l];
        }

    for (int sc = 0; sc < 3; ++sc)
        for (int l = 0; l < 3; ++l) {
            const int total = couts[sc][l] * cinps[sc][l];
            fold_layer_kernel<<<(total + 255) / 256, 256, 0, stream>>>(
                PARAM(sc, l, 0), PARAM(sc, l, 1), PARAM(sc, l, 2),
                PARAM(sc, l, 3), PARAM(sc, l, 4), PARAM(sc, l, 5),
                couts[sc][l], cins[sc][l], cinps[sc][l], Wp[sc][l], Bp[sc][l]);
        }

    float* out_xyz = (float*)d_out;                         // [8,3,512]
    float* out_pts = out_xyz + (size_t)BATCH * 3 * SOUT;    // [8,320,512]

    fps_kernel<<<BATCH, 256, 0, stream>>>(xyz, fps_idx, nxyz, out_xyz);

    const int ngroups = BATCH * SOUT;
    msg_group_kernel<16, 32, 32, 64, 32, 64><<<ngroups, 128, 0, stream>>>(
        xyz, pts, nxyz, Wp[0][0], Bp[0][0], Bp[0][1], Bp[0][2],
        0.1f, out_pts, 0);
    msg_group_kernel<32, 64, 64, 128, 64, 128><<<ngroups, 128, 0, stream>>>(
        xyz, pts, nxyz, Wp[1][0], Bp[1][0], Bp[1][1], Bp[1][2],
        0.2f, out_pts, 64);
    msg_group_kernel<128, 64, 96, 128, 96, 128><<<ngroups, 128, 0, stream>>>(
        xyz, pts, nxyz, Wp[2][0], Bp[2][0], Bp[2][1], Bp[2][2],
        0.4f, out_pts, 192);
}
